// MARN1_nsps_47863115546630
// MI455X (gfx1250) — matrix operand order repaired and checked by reading, not compiled, not run on hardware
//
#include <hip/hip_runtime.h>
#include <math.h>

// ======================= CDNA5 WMMA types & helpers =======================
typedef __attribute__((ext_vector_type(16))) __bf16 v16bf;
typedef __attribute__((ext_vector_type(8)))  float  v8f;

struct U32B { uint4 a, b; };   // 32B container to assemble a v16bf fragment

__device__ __forceinline__ unsigned short f2bf_bits(float f) {
  unsigned u = __builtin_bit_cast(unsigned, f);
  unsigned r = u + 0x7FFFu + ((u >> 16) & 1u);     // round-to-nearest-even
  return (unsigned short)(r >> 16);
}
__device__ __forceinline__ float sigm(float x){ return 1.f/(1.f+__expf(-x)); }

// ======================= LDS-tiled strided-batched WMMA GEMM ==============
// C[b1,b2,m,n] = act( alpha * sum_k A[m,k]*B[k,n] + bias[n] + Res[...] )
// A fp32 row-major (lda); B fp32 row-major KxN (transB=0) or NxK (transB=1).
// Block (8 waves) computes a 64x128 tile: waves in 4(M) x 2(N) grid, each wave
// owns 16x64 -> four v_wmma_f32_16x16x32_bf16 per 32-wide K chunk.
// Software pipelined: global loads for chunk k+1 are issued before the WMMAs
// of chunk k (branch-free clamped addressing); all 8 B-fragment ds_load_b128
// are issued before the WMMA chain so dscnt waits are partial. Fragment
// layouts per ISA 7.12.2 (wave32):
//   A lane frag: K in [8g,8g+7] and [16+8g,16+8g+7]  -> two 16B chunks
//   B lane frag: K in [8g,8g+7] and [16+8g,16+8g+7]  -> two 16B chunks
#define LDSK 40   // padded row length (halves): 80B, 16B aligned, staggers banks

__global__ void k_gemm(const float* __restrict__ A, const float* __restrict__ B,
                       float* __restrict__ C, const float* __restrict__ bias,
                       const float* __restrict__ Res,
                       int M,int N,int K,int lda,int ldb,int ldc,int ldr,
                       long long aS1,long long bS1,long long cS1,long long rS1,int nB1,
                       long long aS2,long long bS2,long long cS2,long long rS2,int nB2,
                       int transB,int act,float alpha)
{
  __shared__ unsigned short As[64][LDSK];
  __shared__ unsigned short Bs[128][LDSK];

  const int tMB = (M + 63) >> 6, tNB = (N + 127) >> 7;
  long long bid = blockIdx.x;
  int tn = (int)(bid % tNB); bid /= tNB;
  int tm = (int)(bid % tMB); bid /= tMB;
  int b2 = (int)(bid % nB2); int b1 = (int)(bid / nB2);

  const float* Ap = A + (long long)b1*aS1 + (long long)b2*aS2;
  const float* Bp = B + (long long)b1*bS1 + (long long)b2*bS2;
  float*       Cp = C + (long long)b1*cS1 + (long long)b2*cS2;

  const int m0 = tm << 6, n0 = tn << 7;
  const int t    = threadIdx.x;
  const int lane = t & 31;
  const int wv   = t >> 5;            // 0..7
  const int wm   = wv >> 1;           // 0..3 : wave row (16 rows each)
  const int wn   = wv & 1;            // 0..1 : wave col (64 cols each)
  const int l16  = lane & 15, g = lane >> 4;

  const v8f z8 = {0.f,0.f,0.f,0.f,0.f,0.f,0.f,0.f};
  v8f acc[4] = {z8, z8, z8, z8};

  const int akj = t & 31,  ami = t >> 5;    // A staging: 32 k x 8 m (x8 iters)
  const int bn0 = t & 127, bk0 = t >> 7;    // B staging transB=0: 128 n x 2 k (x16)
  const int bk1 = t & 31,  bn1 = t >> 5;    // B staging transB=1: 32 k x 8 n (x16)

  float av[8], bv[16];
  // issue all global loads for K-chunk kc into registers (clamped, branch-free)
  auto loadRegs = [&](int kc){
#pragma unroll
    for (int i = 0; i < 8; ++i) {
      int gm = m0 + ami + 8*i, gk = kc + akj;
      bool ok = (gm < M) && (gk < K);
      long long off = (long long)gm*lda + gk;
      float v = Ap[ok ? off : 0];
      av[i] = ok ? v : 0.f;
    }
    if (!transB) {
#pragma unroll
      for (int i = 0; i < 16; ++i) {
        int gk = kc + bk0 + 2*i, gn = n0 + bn0;
        bool ok = (gk < K) && (gn < N);
        long long off = (long long)gk*ldb + gn;
        float v = Bp[ok ? off : 0];
        bv[i] = ok ? v : 0.f;
      }
    } else {
#pragma unroll
      for (int i = 0; i < 16; ++i) {
        int gn = n0 + bn1 + 8*i, gk = kc + bk1;
        bool ok = (gk < K) && (gn < N);
        long long off = (long long)gn*ldb + gk;
        float v = Bp[ok ? off : 0];
        bv[i] = ok ? v : 0.f;
      }
    }
    // prefetch one chunk ahead in the A stream (global_prefetch_b8)
    int pk = kc + 32;
    int gm = m0 + ami;
    bool okp = (gm < M) && (pk < K);
    __builtin_prefetch(Ap + (long long)gm*lda + (okp ? pk : 0), 0, 1);
  };

  loadRegs(0);
  for (int k0 = 0; k0 < K; k0 += 32) {
    // ---- commit staged registers to LDS (bf16) ----
#pragma unroll
    for (int i = 0; i < 8; ++i) As[ami + 8*i][akj] = f2bf_bits(av[i]);
    if (!transB) {
#pragma unroll
      for (int i = 0; i < 16; ++i) Bs[bn0][bk0 + 2*i] = f2bf_bits(bv[i]);
    } else {
#pragma unroll
      for (int i = 0; i < 16; ++i) Bs[bn1 + 8*i][bk1] = f2bf_bits(bv[i]);
    }
    __syncthreads();

    // ---- issue next chunk's global loads (overlap with WMMAs below) ----
    if (k0 + 32 < K) loadRegs(k0 + 32);

    // ---- all fragment LDS reads first (partial dscnt waits), then WMMAs ----
    const int arow = (wm << 4) + l16;
    U32B au;
    au.a = *(const uint4*)&As[arow][8*g];
    au.b = *(const uint4*)&As[arow][16 + 8*g];
    v16bf af = __builtin_bit_cast(v16bf, au);

    v16bf bfj[4];
#pragma unroll
    for (int j = 0; j < 4; ++j) {
      int bcol = (wn << 6) + (j << 4) + l16;
      U32B bu;
      bu.a = *(const uint4*)&Bs[bcol][8*g];
      bu.b = *(const uint4*)&Bs[bcol][16 + 8*g];
      bfj[j] = __builtin_bit_cast(v16bf, bu);
    }
#pragma unroll
    for (int j = 0; j < 4; ++j) {
      acc[j] = __builtin_amdgcn_wmma_f32_16x16x32_bf16(false, af, false, bfj[j],
                                                       (short)0, acc[j], false, false);
    }
    __syncthreads();
  }

  // ---- epilogue: bias / residual / relu, bounds-checked stores ----
#pragma unroll
  for (int j = 0; j < 4; ++j) {
    int cn = n0 + (wn << 6) + (j << 4) + l16;
    if (cn < N) {
#pragma unroll
      for (int v = 0; v < 8; ++v) {
        int cm = m0 + (wm << 4) + v + (g << 3);
        if (cm < M) {
          float val = acc[j][v] * alpha;
          if (bias) val += bias[cn];
          if (Res)  val += Res[(long long)b1*rS1 + (long long)b2*rS2 + (long long)cm*ldr + cn];
          if (act == 1) val = fmaxf(val, 0.f);
          Cp[(long long)cm*ldc + cn] = val;
        }
      }
    }
  }
}

// ======================= row kernels (wave per row) =======================
__global__ void k_layernorm(const float* __restrict__ X, float* __restrict__ Y,
                            const float* __restrict__ g, const float* __restrict__ b,
                            int R, int L, float eps) {
  int lane = threadIdx.x & 31;
  int row = blockIdx.x * (blockDim.x >> 5) + (threadIdx.x >> 5);
  if (row >= R) return;
  const float* x = X + (long long)row * L;
  float s = 0.f, s2 = 0.f;
  for (int i = lane; i < L; i += 32) { float v = x[i]; s += v; s2 += v*v; }
  for (int o = 16; o; o >>= 1) { s += __shfl_xor(s, o, 32); s2 += __shfl_xor(s2, o, 32); }
  float mu = s / L;
  float inv = rsqrtf(s2 / L - mu*mu + eps);
  float* y = Y + (long long)row * L;
  for (int i = lane; i < L; i += 32) y[i] = (x[i]-mu)*inv*g[i] + b[i];
}

__global__ void k_softmax(const float* __restrict__ X, float* __restrict__ Y,
                          long long R, int L) {
  int lane = threadIdx.x & 31;
  long long row = (long long)blockIdx.x * (blockDim.x >> 5) + (threadIdx.x >> 5);
  if (row >= R) return;
  const float* x = X + row * L;
  float m = -3.4e38f;
  for (int i = lane; i < L; i += 32) m = fmaxf(m, x[i]);
  for (int o = 16; o; o >>= 1) m = fmaxf(m, __shfl_xor(m, o, 32));
  float s = 0.f;
  for (int i = lane; i < L; i += 32) s += __expf(x[i]-m);
  for (int o = 16; o; o >>= 1) s += __shfl_xor(s, o, 32);
  float inv = 1.f / s;
  float* y = Y + row * L;
  for (int i = lane; i < L; i += 32) y[i] = __expf(x[i]-m)*inv;
}

// ======================= elementwise helpers ==============================
__global__ void k_zero(float* p, long long n){
  long long i = (long long)blockIdx.x*blockDim.x + threadIdx.x; if (i < n) p[i] = 0.f; }
__global__ void k_copy(const float* s, float* d, long long n){
  long long i = (long long)blockIdx.x*blockDim.x + threadIdx.x; if (i < n) d[i] = s[i]; }
__global__ void k_add(const float* a, const float* b, float* c, long long n){
  long long i = (long long)blockIdx.x*blockDim.x + threadIdx.x; if (i < n) c[i] = a[i]+b[i]; }
__global__ void k_add4(const float* a,const float* b,const float* c,const float* d,float* o,int n){
  int i = blockIdx.x*blockDim.x + threadIdx.x; if (i < n) o[i] = a[i]+b[i]+c[i]+d[i]; }
__global__ void k_copy2d(const float* src, float* dst, int R, int sLd, int sOff,
                         int dLd, int dOff, int L){
  long long i = (long long)blockIdx.x*blockDim.x + threadIdx.x;
  if (i >= (long long)R*L) return;
  int j = (int)(i % L); long long r = i / L;
  dst[r*dLd + dOff + j] = src[r*sLd + sOff + j];
}

// ======================= sequence reverse =================================
__global__ void k_lengths(const float* um, int* len, int B, int T){
  int b = blockIdx.x*blockDim.x + threadIdx.x; if (b >= B) return;
  float s = 0.f; for (int t = 0; t < T; ++t) s += um[b*T + t];
  len[b] = (int)(s + 0.5f);
}
__global__ void k_reverse(const float* X, float* Y, const int* len, int T, int B, int L){
  long long i = (long long)blockIdx.x*blockDim.x + threadIdx.x;
  if (i >= (long long)T*B*L) return;
  int j = (int)(i % L); long long r = i / L;
  int b = (int)(r % B); int t = (int)(r / B);
  int lb = len[b];
  Y[i] = (t < lb) ? X[((long long)(lb-1-t)*B + b)*L + j] : 0.f;
}

// ======================= MARN scan kernels ================================
__global__ void k_prepare(const float* qm, const float* q, float* qs0, float* ql0, int B){
  int i = blockIdx.x*blockDim.x + threadIdx.x; if (i >= B*128) return;
  int b = i >> 7, j = i & 127;
  int idx = (qm[b*2+1] > qm[b*2+0]) ? 1 : 0;
  qs0[i] = q[((long long)b*2 + idx)*128 + j];
  ql0[i] = q[((long long)b*2 + (1-idx))*128 + j];
}
__global__ void k_gru_gate(const float* gi, const float* gh, const float* qs0,
                           const float* qm, const float* ql0, float* q, float* hs, int B){
  int i = blockIdx.x*blockDim.x + threadIdx.x; if (i >= B*128) return;
  int b = i >> 7, j = i & 127;
  float r = sigm(gi[b*384+j]       + gh[b*384+j]);
  float z = sigm(gi[b*384+128+j]   + gh[b*384+128+j]);
  float n = tanhf(gi[b*384+256+j] + r*gh[b*384+256+j]);
  float h = (1.f-z)*n + z*qs0[i];
  hs[i] = h;
  float q0 = qm[b*2+0], q1 = qm[b*2+1];
  q[((long long)b*2+0)*128+j] = ql0[i]*(1.f-q0) + h*q0;
  q[((long long)b*2+1)*128+j] = ql0[i]*(1.f-q1) + h*q1;
}
__global__ void k_build_cat(const float* x100, const float* h, const float* z,
                            const float* s, float* cat, int B){
  int i = blockIdx.x*blockDim.x + threadIdx.x; if (i >= B*484) return;
  int b = i/484, j = i%484;
  float v;
  if      (j < 100) v = x100[b*100 + j];
  else if (j < 228) v = h[b*128 + (j-100)];
  else if (j < 356) v = z[b*128 + (j-228)];
  else              v = s[b*128 + (j-356)];
  cat[i] = v;
}
__global__ void k_lsthm_gate(const float* s, float* c, float* h, float* hout, int B){
  int i = blockIdx.x*blockDim.x + threadIdx.x; if (i >= B*128) return;
  int b = i >> 7, j = i & 127;
  const float* sb = s + (long long)b*512;
  float f  = sigm(sb[j]);
  float ii = sigm(sb[128+j]);
  float o  = sigm(sb[256+j]);
  float ch = tanhf(sb[384+j]);
  float ct = f*c[i] + ii*ch;
  c[i] = ct;
  float hv = tanhf(ct)*o;
  h[i] = hv; hout[i] = hv;
}
// z_l[b,q] = sum_k softmax_k( (c_l[b,q]*s_b) * Wk[k] ) * c_a[b,k]; s_b = c_a.Wq/sqrt(128)
__global__ void k_ca_small(const float* c_l, const float* c_a, const float* Wq,
                           const float* Wk, float* z, int B){
  __shared__ float sWk[128], sCa[128], red[128];
  int b = blockIdx.x, tid = threadIdx.x;
  sWk[tid] = Wk[tid];
  float ca = c_a[b*128 + tid];
  sCa[tid] = ca;
  red[tid] = ca * Wq[tid];
  __syncthreads();
  for (int o = 64; o; o >>= 1) { if (tid < o) red[tid] += red[tid+o]; __syncthreads(); }
  float s = red[0] * 0.08838834764831845f;
  float tq = c_l[b*128 + tid] * s;
  float m = -3.4e38f;
  for (int k = 0; k < 128; ++k) m = fmaxf(m, tq*sWk[k]);
  float den = 0.f, num = 0.f;
  for (int k = 0; k < 128; ++k){ float e = __expf(tq*sWk[k]-m); den += e; num += e*sCa[k]; }
  z[b*128 + tid] = num/den;
}

// ======================= final stage ======================================
__global__ void k_final_z(const float* hfl, const float* hbl, const float* at2,
                          const float* hfa, const float* hba, const float* at1,
                          const float* resid, const float* w, float* z, long long R){
  long long i = (long long)blockIdx.x*blockDim.x + threadIdx.x;
  if (i >= R*712) return;
  long long r = i / 712; int j = (int)(i % 712);
  float w0 = w[0], w1 = w[1], v;
  if      (j < 128) v = w0*hfl[r*128 + j];
  else if (j < 256) v = w0*hbl[r*128 + (j-128)];
  else if (j < 356) v = w0*at2[r*100 + (j-256)];
  else if (j < 484) v = w1*hfa[r*128 + (j-356)];
  else if (j < 612) v = w1*hba[r*128 + (j-484)];
  else              v = w1*at1[r*100 + (j-612)];
  z[i] = v + resid[i];
}
__global__ void k_logsoftmax_out(const float* X, float* out, int T, int B, int C){
  int r = blockIdx.x*blockDim.x + threadIdx.x;
  if (r >= T*B) return;
  int b = r % B, t = r / B;
  const float* x = X + (long long)r*C;
  float m = -3.4e38f; for (int c = 0; c < C; ++c) m = fmaxf(m, x[c]);
  float s = 0.f;      for (int c = 0; c < C; ++c) s += __expf(x[c]-m);
  float ls = logf(s);
  float* o = out + ((long long)b*T + t)*C;
  for (int c = 0; c < C; ++c) o[c] = x[c]-m-ls;
}

// ======================= host-side launchers ==============================
static void gemmB(hipStream_t st, const float* A, const float* B, float* C,
  const float* bias, const float* Res,
  int M,int N,int K,int lda,int ldb,int ldc,int ldr,
  long long aS1,long long bS1,long long cS1,long long rS1,int nB1,
  long long aS2,long long bS2,long long cS2,long long rS2,int nB2,
  int transB,int act,float alpha)
{
  long long blocks = (long long)((M+63)/64)*((N+127)/128)*nB1*nB2;
  k_gemm<<<(unsigned)blocks,256,0,st>>>(A,B,C,bias,Res,M,N,K,lda,ldb,ldc,ldr,
    aS1,bS1,cS1,rS1,nB1,aS2,bS2,cS2,rS2,nB2,transB,act,alpha);
}
static void gemm1(hipStream_t st, const float* A, const float* B, float* C,
  const float* bias, const float* Res,
  int M,int N,int K,int lda,int ldb,int ldc,int ldr,int transB,int act,float alpha)
{ gemmB(st,A,B,C,bias,Res,M,N,K,lda,ldb,ldc,ldr,0,0,0,0,1,0,0,0,0,1,transB,act,alpha); }

extern "C" void kernel_launch(void* const* d_in, const int* in_sizes, int n_in,
                              void* d_out, int out_size, void* d_ws, size_t ws_size,
                              hipStream_t stream)
{
  (void)in_sizes; (void)n_in; (void)out_size; (void)ws_size;
  const int T = 128, Bc = 64, DL = 100;
  const int R = T*Bc;                      // 8192 (t*B+b "TB" layout everywhere)
  auto P = [&](int i){ return (const float*)d_in[i]; };

  // inputs: x, qmask, umask, then params flattened with sorted dict keys
  const float* x     = P(0);
  const float* qmask = P(1);
  const float* umask = P(2);
  struct Ca { const float *Wk,*Wq,*Wv,*lnb,*lng; };
  Ca caA{P(3),P(4),P(5),P(6),P(7)};        // ca_a2l
  Ca caL{P(8),P(9),P(10),P(11),P(12)};     // ca_l2a
  struct Cell { const float *caWk,*caWq,*gWhh,*gWih,*gbhh,*gbih,
                 *aS,*aU,*aV,*aW,*abS,*abU,*abV,*abW,
                 *lS,*lU,*lV,*lW,*lbS,*lbU,*lbV,*lbW; };
  auto LC = [&](int b){ Cell c{P(b),P(b+1),P(b+2),P(b+3),P(b+4),P(b+5),
    P(b+6),P(b+7),P(b+8),P(b+9),P(b+10),P(b+11),P(b+12),P(b+13),
    P(b+14),P(b+15),P(b+16),P(b+17),P(b+18),P(b+19),P(b+20),P(b+21)}; return c; };
  Cell cellB = LC(13), cellF = LC(35);
  struct Enc { const float *b1,*b2,*fc,*ln1b,*ln1g,*ln2b,*ln2g,*w1,*w2,*wk,*wq,*wv; };
  auto LE = [&](int b){ Enc e{P(b),P(b+1),P(b+2),P(b+3),P(b+4),P(b+5),P(b+6),
    P(b+7),P(b+8),P(b+9),P(b+10),P(b+11)}; return e; };
  Enc encA = LE(57), encL = LE(69);
  const float* fc_W = P(81); const float* fc_b = P(82);
  const float* in_W = P(83); const float* in_b = P(84);
  const float* o1_W = P(85); const float* o1_b = P(86);
  const float* o2_W = P(87); const float* o2_b = P(88);
  const float* pp   = P(89);

  // workspace bump allocator
  size_t off = 0;
  auto alloc = [&](size_t n)->float*{
    float* p = (float*)((char*)d_ws + off);
    off += ((n*sizeof(float) + 255) & ~(size_t)255);
    return p;
  };
  auto NB = [](long long n){ return (unsigned)((n + 255)/256); };

  float* xl0   = alloc((size_t)R*DL);
  float* xa0   = alloc((size_t)R*DL);
  float* xcat  = alloc((size_t)R*200);
  float* eqQ   = alloc((size_t)R*320);
  float* eqK   = alloc((size_t)R*320);
  float* eqV   = alloc((size_t)R*320);
  float* attL  = alloc((size_t)Bc*8*T*T);
  float* eo    = alloc((size_t)R*320);
  float* tmpA  = alloc((size_t)R*DL);
  float* tmpI  = alloc((size_t)R*40);
  float* encO1 = alloc((size_t)R*DL);
  float* xl1   = alloc((size_t)R*DL);
  float* xa1   = alloc((size_t)R*DL);
  float* encIn = alloc((size_t)R*DL);
  float* xlF   = alloc((size_t)R*DL);
  float* xaF   = alloc((size_t)R*DL);
  int*   lens  = (int*)alloc(Bc);
  float* xcatR = alloc((size_t)R*200);
  float* xlR   = alloc((size_t)R*DL);
  float* xaR   = alloc((size_t)R*DL);
  float* qmR   = alloc((size_t)R*2);
  float* WcL0  = alloc((size_t)484*512); float* WcA0 = alloc((size_t)484*512);
  float* WcL1  = alloc((size_t)484*512); float* WcA1 = alloc((size_t)484*512);
  float* bcL0  = alloc(512); float* bcA0 = alloc(512);
  float* bcL1  = alloc(512); float* bcA1 = alloc(512);
  float* h_l = alloc((size_t)Bc*128); float* h_a = alloc((size_t)Bc*128);
  float* c_l = alloc((size_t)Bc*128); float* c_a = alloc((size_t)Bc*128);
  float* z_l = alloc((size_t)Bc*128);
  float* qst = alloc((size_t)Bc*2*128);
  float* qs0 = alloc((size_t)Bc*128); float* ql0 = alloc((size_t)Bc*128);
  float* hs  = alloc((size_t)Bc*128);
  float* gi  = alloc((size_t)Bc*384); float* gh = alloc((size_t)Bc*384);
  float* catLb = alloc((size_t)Bc*484); float* catAb = alloc((size_t)Bc*484);
  float* sL = alloc((size_t)Bc*512); float* sA = alloc((size_t)Bc*512);
  float* hf_l = alloc((size_t)T*Bc*128); float* hf_a = alloc((size_t)T*Bc*128);
  float* hb_l = alloc((size_t)T*Bc*128); float* hb_a = alloc((size_t)T*Bc*128);
  float* hblR = alloc((size_t)T*Bc*128); float* hbaR = alloc((size_t)T*Bc*128);
  float* cq   = alloc((size_t)R*DL); float* ck = alloc((size_t)R*DL); float* cv = alloc((size_t)R*DL);
  float* attC = alloc((size_t)Bc*T*T);
  float* catt = alloc((size_t)R*DL);
  float* attn1 = alloc((size_t)R*DL); float* attn2 = alloc((size_t)R*DL);
  float* resid = alloc((size_t)R*712);
  float* zb    = alloc((size_t)R*712);
  float* hid   = alloc((size_t)R*32);
  float* cls   = alloc((size_t)R*6);
  float* wbuf  = alloc(2);

  auto layernorm = [&](const float* X_, float* Y_, const float* g_, const float* b_,
                       int Rr, int L, float eps){
    k_layernorm<<<(unsigned)((Rr+7)/8),256,0,stream>>>(X_,Y_,g_,b_,Rr,L,eps);
  };
  auto softmaxRows = [&](float* X_, long long Rr, int L){
    k_softmax<<<(unsigned)((Rr+7)/8),256,0,stream>>>(X_,X_,Rr,L);
  };

  // ---- encoder layer: X (R x 100) -> Y (R x 100), TB layout
  auto encoder = [&](const float* Xin, const Enc& e, float* Yout){
    gemm1(stream, Xin, e.wq, eqQ, nullptr, nullptr, R,320,DL, DL,320,320,0, 0,0,1.f);
    gemm1(stream, Xin, e.wk, eqK, nullptr, nullptr, R,320,DL, DL,320,320,0, 0,0,1.f);
    gemm1(stream, Xin, e.wv, eqV, nullptr, nullptr, R,320,DL, DL,320,320,0, 0,0,1.f);
    // logits[b,h,i,j] = q.k / sqrt(40)   (batch dims: b then h)
    gemmB(stream, eqQ, eqK, attL, nullptr, nullptr,
          T,T,40, Bc*320,Bc*320,T,0,
          320,320,(long long)8*T*T,0,Bc,
          40,40,(long long)T*T,0,8,
          1,0, 0.15811388300841897f);
    softmaxRows(attL, (long long)Bc*8*T, T);
    // o[i,b,h,d] = attn . v
    gemmB(stream, attL, eqV, eo, nullptr, nullptr,
          T,40,T, T,Bc*320,Bc*320,0,
          (long long)8*T*T,320,320,0,Bc,
          (long long)T*T,40,40,0,8,
          0,0,1.f);
    gemm1(stream, eo, e.fc, tmpA, nullptr, Xin, R,DL,320, 320,DL,DL,DL, 0,0,1.f);
    layernorm(tmpA, encO1, e.ln1g, e.ln1b, R, DL, 1e-5f);
    gemm1(stream, encO1, e.w1, tmpI, e.b1, nullptr, R,40,DL, DL,40,40,0, 0,1,1.f);
    gemm1(stream, tmpI,  e.w2, tmpA, e.b2, encO1,   R,DL,40, 40,DL,DL,DL, 0,0,1.f);
    layernorm(tmpA, Yout, e.ln2g, e.ln2b, R, DL, 1e-5f);
  };

  // ---- cross attention 2 (per-b attention over time)
  auto crossatt = [&](const float* X1, const float* X2, const Ca& p, float* Yout){
    gemm1(stream, X1, p.Wq, cq, nullptr, nullptr, R,DL,DL, DL,DL,DL,0, 0,0,1.f);
    gemm1(stream, X2, p.Wk, ck, nullptr, nullptr, R,DL,DL, DL,DL,DL,0, 0,0,1.f);
    gemm1(stream, X2, p.Wv, cv, nullptr, nullptr, R,DL,DL, DL,DL,DL,0, 0,0,1.f);
    gemmB(stream, cq, ck, attC, nullptr, nullptr,
          T,T,DL, Bc*DL,Bc*DL,T,0,
          DL,DL,(long long)T*T,0,Bc,
          0,0,0,0,1,
          1,0, 0.1f);
    softmaxRows(attC, (long long)Bc*T, T);
    gemmB(stream, attC, cv, catt, nullptr, X1,
          T,DL,T, T,Bc*DL,Bc*DL,Bc*DL,
          (long long)T*T,DL,DL,DL,Bc,
          0,0,0,0,1,
          0,0,1.f);
    layernorm(catt, Yout, p.lng, p.lnb, R, DL, 1e-6f);
  };

  // ---- MARN scan (one direction)
  auto runscan = [&](const float* xc, const float* xls, const float* xas, const float* qms,
                     const Cell& cp, float* WcL, float* WcA, float* bcL, float* bcA,
                     float* houtL, float* houtA){
    // concat LSTHM weights once: rows [W(100); U(128); V(128); S(128)] x 512
    k_copy<<<NB(100*512),256,0,stream>>>(cp.lW, WcL,            100*512);
    k_copy<<<NB(128*512),256,0,stream>>>(cp.lU, WcL+100*512,    128*512);
    k_copy<<<NB(128*512),256,0,stream>>>(cp.lV, WcL+228*512,    128*512);
    k_copy<<<NB(128*512),256,0,stream>>>(cp.lS, WcL+356*512,    128*512);
    k_add4<<<NB(512),256,0,stream>>>(cp.lbW,cp.lbU,cp.lbV,cp.lbS,bcL,512);
    k_copy<<<NB(100*512),256,0,stream>>>(cp.aW, WcA,            100*512);
    k_copy<<<NB(128*512),256,0,stream>>>(cp.aU, WcA+100*512,    128*512);
    k_copy<<<NB(128*512),256,0,stream>>>(cp.aV, WcA+228*512,    128*512);
    k_copy<<<NB(128*512),256,0,stream>>>(cp.aS, WcA+356*512,    128*512);
    k_add4<<<NB(512),256,0,stream>>>(cp.abW,cp.abU,cp.abV,cp.abS,bcA,512);
    // zero carries
    k_zero<<<NB(Bc*128),256,0,stream>>>(h_l, Bc*128);
    k_zero<<<NB(Bc*128),256,0,stream>>>(h_a, Bc*128);
    k_zero<<<NB(Bc*128),256,0,stream>>>(c_l, Bc*128);
    k_zero<<<NB(Bc*128),256,0,stream>>>(c_a, Bc*128);
    k_zero<<<NB(Bc*128),256,0,stream>>>(z_l, Bc*128);
    k_zero<<<NB(Bc*256),256,0,stream>>>(qst, Bc*256);
    for (int t = 0; t < T; ++t) {
      const float* qmt = qms + (size_t)t*Bc*2;
      k_prepare<<<NB(Bc*128),256,0,stream>>>(qmt, qst, qs0, ql0, Bc);
      gemm1(stream, xc + (size_t)t*Bc*200, cp.gWih, gi, cp.gbih, nullptr,
            Bc,384,200, 200,384,384,0, 0,0,1.f);
      gemm1(stream, qs0, cp.gWhh, gh, cp.gbhh, nullptr,
            Bc,384,128, 128,384,384,0, 0,0,1.f);
      k_gru_gate<<<NB(Bc*128),256,0,stream>>>(gi, gh, qs0, qmt, ql0, qst, hs, Bc);
      k_build_cat<<<NB(Bc*484),256,0,stream>>>(xls + (size_t)t*Bc*100, h_l, z_l, hs, catLb, Bc);
      k_build_cat<<<NB(Bc*484),256,0,stream>>>(xas + (size_t)t*Bc*100, h_a, z_l, hs, catAb, Bc);
      gemm1(stream, catLb, WcL, sL, bcL, nullptr, Bc,512,484, 484,512,512,0, 0,0,1.f);
      gemm1(stream, catAb, WcA, sA, bcA, nullptr, Bc,512,484, 484,512,512,0, 0,0,1.f);
      k_lsthm_gate<<<NB(Bc*128),256,0,stream>>>(sL, c_l, h_l, houtL + (size_t)t*Bc*128, Bc);
      k_lsthm_gate<<<NB(Bc*128),256,0,stream>>>(sA, c_a, h_a, houtA + (size_t)t*Bc*128, Bc);
      k_ca_small<<<Bc,128,0,stream>>>(c_l, c_a, cp.caWq, cp.caWk, z_l, Bc);
    }
  };

  // ======================= forward pass ==================================
  // input projection (x rows are already t*B+b with ld=1124)
  gemm1(stream, x, in_W, xl0, in_b, nullptr, R,DL,1024, 1124,DL,DL,0, 0,0,1.f);
  k_copy2d<<<NB((long long)R*DL),256,0,stream>>>(x, xa0, R, 1124,1024, DL,0, DL);
  k_copy2d<<<NB((long long)R*DL),256,0,stream>>>(xl0, xcat, R, DL,0, 200,0,   DL);
  k_copy2d<<<NB((long long)R*DL),256,0,stream>>>(xa0, xcat, R, DL,0, 200,DL,  DL);

  // encoders (layer params shared between the two applications)
  encoder(xl0, encL, xl1);
  encoder(xa0, encA, xa1);
  k_add<<<NB((long long)R*DL),256,0,stream>>>(xl0, xl1, encIn, (long long)R*DL);
  encoder(encIn, encL, xlF);
  k_add<<<NB((long long)R*DL),256,0,stream>>>(xa0, xa1, encIn, (long long)R*DL);
  encoder(encIn, encA, xaF);

  // sequence reversal inputs
  k_lengths<<<1,64,0,stream>>>(umask, lens, Bc, T);
  k_reverse<<<NB((long long)R*200),256,0,stream>>>(xcat, xcatR, lens, T, Bc, 200);
  k_reverse<<<NB((long long)R*DL),256,0,stream>>>(xlF, xlR, lens, T, Bc, DL);
  k_reverse<<<NB((long long)R*DL),256,0,stream>>>(xaF, xaR, lens, T, Bc, DL);
  k_reverse<<<NB((long long)R*2),256,0,stream>>>(qmask, qmR, lens, T, Bc, 2);

  // bidirectional MARN scans
  runscan(xcat,  xlF, xaF, qmask, cellF, WcL0, WcA0, bcL0, bcA0, hf_l, hf_a);
  runscan(xcatR, xlR, xaR, qmR,   cellB, WcL1, WcA1, bcL1, bcA1, hb_l, hb_a);
  k_reverse<<<NB((long long)R*128),256,0,stream>>>(hb_l, hblR, lens, T, Bc, 128);
  k_reverse<<<NB((long long)R*128),256,0,stream>>>(hb_a, hbaR, lens, T, Bc, 128);

  // cross attentions
  crossatt(xlF, xaF, caL, attn1);   // ca_l2a
  crossatt(xaF, xlF, caA, attn2);   // ca_a2l

  // modality weights + residual + head
  k_softmax<<<1,256,0,stream>>>(pp, wbuf, 1, 2);
  gemm1(stream, xlF, fc_W, resid, fc_b, nullptr, R,712,DL, DL,712,712,0, 0,1,1.f);
  k_final_z<<<NB((long long)R*712),256,0,stream>>>(hf_l, hblR, attn2, hf_a, hbaR, attn1,
                                                   resid, wbuf, zb, R);
  gemm1(stream, zb,  o1_W, hid, o1_b, nullptr, R,32,712, 712,32,32,0, 0,1,1.f);
  gemm1(stream, hid, o2_W, cls, o2_b, nullptr, R,6,32,   32,6,6,0,   0,0,1.f);
  k_logsoftmax_out<<<NB(R),256,0,stream>>>(cls, (float*)d_out, T, Bc, 6);
}
